// CrossAttention_63608465654184
// MI455X (gfx1250) — matrix operand order repaired and checked by reading, not compiled, not run on hardware
//
#include <hip/hip_runtime.h>

// ---------------------------------------------------------------------------
// Fused windowed self+cross attention for MI455X (gfx1250, wave32, WMMA).
// One workgroup (256 thr = 8 waves) per batch window b (1024 windows).
// All GEMMs via v_wmma_f32_16x16x32_f16; activations staged in LDS (304KB),
// weights pre-transposed+converted to f16 in d_ws (L2 resident).
// v/v1 stored dim-major so attn@v B-fragments are contiguous v16h LDS loads.
// 2x N-tile blocking halves A-fragment LDS traffic (LDS-BW limited otherwise).
// ---------------------------------------------------------------------------

typedef _Float16 h16;
typedef __attribute__((ext_vector_type(16))) _Float16 v16h;
typedef __attribute__((ext_vector_type(8)))  float    v8f;
typedef __attribute__((ext_vector_type(4)))  _Float16 v4h;
typedef __attribute__((ext_vector_type(4)))  float    v4f;

#define L_TOK   64
#define CDIM    512
#define NHEAD   16
#define HDIM    32
#define QSCALE  0.17677669529663687f   // 1/sqrt(32)

// LDS arena byte offsets (total 311296 B = 304 KB <= 320 KB/WGP)
#define OFF_R0   0        // 64KB : x16 / xr (tok-major)  -> k1 / x1 (head-major)
#define OFF_Q    65536    // 64KB : qkv.q (head-major)    -> h (tok-major)
#define OFF_K    131072   // 64KB : qkv.k (head-major)    -> y16/z16 (tok) -> q1 (head)
#define OFF_V    196608   // 64KB : qkv.v (dim-major)     -> v1 (dim-major)
#define OFF_SCR  262144   // 8 waves * 6144 B (4KB f32 scores + 2KB f16 probs)

// ---- WMMA fragment helpers (A/B use contiguous k = 16*g + s, wave32) ----------

// 16x32 f16 A fragment: lane m = lane&15, element s = K 16*g + s (contiguous)
__device__ __forceinline__ v16h ldA(const h16* __restrict__ base, int rs, int m,
                                    int kb, int g) {
  v16h a;
#pragma unroll
  for (int v = 0; v < 8; ++v) {
    int k0 = kb + 2 * v + 16 * g;
    const h16* p = base + m * rs + k0;
    a[2 * v]     = p[0];
    a[2 * v + 1] = p[1];
  }
  return a;
}

__device__ __forceinline__ v8f wmma_(v16h a, v16h b, v8f c) {
  return __builtin_amdgcn_wmma_f32_16x16x32_f16(false, a, false, b, (short)0, c,
                                                false, false);
}

__device__ __forceinline__ v8f vzero8() {
  v8f z;
#pragma unroll
  for (int i = 0; i < 8; ++i) z[i] = 0.0f;
  return z;
}

// ---- Generic 64x512 = A(64x512) @ WT + bias GEMM, 2x N-tile blocked --------
// AMODE: 0 = A tok-major [64][512], 1 = A head-major [16][64][32]
// OMODE: 0 = f16 LDS tok-major, 1 = f16 LDS head-major,
//        2 = f32 global, 3 = f16 LDS dim-major ([head][dim][token], for v1)
template <int AMODE, int OMODE, bool SC>
__device__ __forceinline__ void gemm512(const h16* __restrict__ A,
                                        const h16* __restrict__ WT,
                                        const float* __restrict__ bias,
                                        h16* ldst, float* gdst, int w, int lane) {
  const int col = lane & 15;
  const int g   = (lane >> 4) & 1;
  for (int p = 0; p < 2; ++p) {
    const int ntA = w + 16 * p;        // owned tiles: pairs (w,w+8),(w+16,w+24)
    v8f acc[2][4];
#pragma unroll
    for (int q2 = 0; q2 < 2; ++q2)
#pragma unroll
      for (int i = 0; i < 4; ++i) acc[q2][i] = vzero8();
    for (int kc = 0; kc < 16; ++kc) {
      v16h bfr0 = *(const v16h*)(WT + (ntA * 16 + col) * 512 + kc * 32 + 16 * g);
      v16h bfr1 = *(const v16h*)(WT + ((ntA + 8) * 16 + col) * 512 + kc * 32 + 16 * g);
#pragma unroll
      for (int mt = 0; mt < 4; ++mt) {
        v16h afr = (AMODE == 0) ? ldA(A, 512, mt * 16 + col, kc * 32, g)
                                : ldA(A + kc * 2048, 32, mt * 16 + col, 0, g);
        acc[0][mt] = wmma_(afr, bfr0, acc[0][mt]);
        acc[1][mt] = wmma_(afr, bfr1, acc[1][mt]);
      }
    }
#pragma unroll
    for (int q2 = 0; q2 < 2; ++q2) {
      const int nt = ntA + 8 * q2;
      float     bb = bias[nt * 16 + col];
#pragma unroll
      for (int mt = 0; mt < 4; ++mt) {
#pragma unroll
        for (int r = 0; r < 8; ++r) {
          int   row = mt * 16 + r + 8 * g;
          float v   = acc[q2][mt][r] + bb;
          if (SC) v *= QSCALE;
          if (OMODE == 0) {
            ldst[row * 512 + nt * 16 + col] = (h16)v;
          } else if (OMODE == 1) {
            int hh = nt >> 1, ph = nt & 1;
            ldst[(hh * 64 + row) * 32 + ph * 16 + col] = (h16)v;
          } else if (OMODE == 3) {
            int hh = nt >> 1, ph = nt & 1;
            ldst[(hh * 32 + ph * 16 + col) * 64 + row] = (h16)v;  // dim-major
          } else {
            gdst[row * 512 + nt * 16 + col] = v;
          }
        }
      }
    }
  }
}

// ---- Per-head attention (scores -> softmax+1 -> @v -> residual) ------------
// q,k head-major [h][tok][dim]; v dim-major [h][dim][tok].
// MODE 0: self-attn, dest = tok-major xr (in-place over x16 in R0)
// MODE 1: cross-attn, dest = head-major x1 (in-place over k1 in R0),
//         residual from tok-major hbuf
template <int MODE>
__device__ __forceinline__ void attn_heads(const h16* __restrict__ qall,
                                           const h16* __restrict__ kall,
                                           const h16* __restrict__ vall,
                                           h16* dst, const h16* __restrict__ hbuf,
                                           float* scr, h16* prob, int w, int lane) {
  const int col = lane & 15;
  const int g   = (lane >> 4) & 1;
  for (int hi = 0; hi < 2; ++hi) {
    int         h  = 2 * w + hi;
    const h16*  qh = qall + h * 2048;
    const h16*  kh = kall + h * 2048;
    const h16*  vh = vall + h * 2048;
    v8f accO[4][2];  // deferred AV outputs so k1->x1 overwrite is safe
#pragma unroll
    for (int mt = 0; mt < 4; ++mt)
#pragma unroll
      for (int n2 = 0; n2 < 2; ++n2) accO[mt][n2] = vzero8();

    for (int mt = 0; mt < 4; ++mt) {
      // scores S(16x64) = q_blk(16x32, pre-scaled) @ k^T
      v16h aq = ldA(qh, 32, mt * 16 + col, 0, g);
#pragma unroll
      for (int nt = 0; nt < 4; ++nt) {
        v16h bk2 = *(const v16h*)(kh + (nt * 16 + col) * 32 + 16 * g);
        v8f  s   = wmma_(aq, bk2, vzero8());
#pragma unroll
        for (int r = 0; r < 8; ++r)
          scr[(r + 8 * g) * 64 + nt * 16 + col] = s[r];
      }
      // softmax_plus_one over each 64-wide row (lane handles half a row)
      {
        int        rr = lane & 15;
        int        hf = lane >> 4;
        const v4f* s4 = (const v4f*)(scr + rr * 64 + hf * 32);
        float      vals[32];
        float      mx = -3.4e38f;
#pragma unroll
        for (int jj = 0; jj < 8; ++jj) {
          v4f vv = s4[jj];
#pragma unroll
          for (int e = 0; e < 4; ++e) {
            vals[jj * 4 + e] = vv[e];
            mx               = fmaxf(mx, vv[e]);
          }
        }
        mx        = fmaxf(mx, __shfl_xor(mx, 16, 32));
        float sum = 0.0f;
#pragma unroll
        for (int j = 0; j < 32; ++j) {
          vals[j] = __expf(vals[j] - mx);
          sum += vals[j];
        }
        sum += __shfl_xor(sum, 16, 32);
        float inv = 1.0f / (sum + 1.0f);
        v4h* p4   = (v4h*)(prob + rr * 64 + hf * 32);
#pragma unroll
        for (int jj = 0; jj < 8; ++jj) {
          v4h hv;
#pragma unroll
          for (int e = 0; e < 4; ++e) hv[e] = (h16)(vals[jj * 4 + e] * inv);
          p4[jj] = hv;
        }
      }
      // out_blk(16x32) += P(16x64) @ v(64x32); v is dim-major -> contiguous B
#pragma unroll
      for (int n2 = 0; n2 < 2; ++n2) {
#pragma unroll
        for (int kc = 0; kc < 2; ++kc) {
          v16h ap  = ldA(prob, 64, col, kc * 32, g);
          v16h bv2 = *(const v16h*)(vh + (n2 * 16 + col) * 64 + kc * 32 + 16 * g);
          accO[mt][n2] = wmma_(ap, bv2, accO[mt][n2]);
        }
      }
    }
    // residual writes (all reads of kh/vh for this head are done)
#pragma unroll
    for (int mt = 0; mt < 4; ++mt) {
#pragma unroll
      for (int n2 = 0; n2 < 2; ++n2) {
#pragma unroll
        for (int r = 0; r < 8; ++r) {
          int row = mt * 16 + r + 8 * g;
          if (MODE == 0) {
            int   c  = h * 32 + n2 * 16 + col;
            float xv = (float)dst[row * 512 + c];
            dst[row * 512 + c] = (h16)(xv + accO[mt][n2][r]);
          } else {
            int   p  = n2 * 16 + col;
            float hv = (float)hbuf[row * 512 + h * 32 + p];
            dst[(h * 64 + row) * 32 + p] = (h16)(hv + accO[mt][n2][r]);
          }
        }
      }
    }
  }
}

// ---- f32 global -> f16 LDS tok-major (64x512) ------------------------------
__device__ __forceinline__ void load_conv(const float* __restrict__ src,
                                          h16* dst, int tid) {
  for (int it = 0; it < 32; ++it) {
    int idx4 = tid + it * 256;
    v4f vv   = ((const v4f*)src)[idx4];
    v4h hv;
#pragma unroll
    for (int j = 0; j < 4; ++j) hv[j] = (h16)vv[j];
    ((v4h*)dst)[idx4] = hv;
  }
}

// ---- weight prep: W(K,N) f32 -> WT(N,K) f16 --------------------------------
__global__ void wprep_kernel(const float* __restrict__ W, h16* __restrict__ WT,
                             int K, int N) {
  int t = blockIdx.x * 256 + threadIdx.x;
  if (t >= K * N) return;
  int n = t / K, k = t - n * K;
  WT[t] = (h16)W[(size_t)k * N + n];
}

// ---- main fused kernel -----------------------------------------------------
__global__ __launch_bounds__(256, 1) void fused_mwa_kernel(
    const float* __restrict__ y, const float* __restrict__ z,
    const float* __restrict__ x, const float* __restrict__ bqkv,
    const float* __restrict__ bq, const float* __restrict__ bk,
    const float* __restrict__ bv, const float* __restrict__ bp,
    const h16* __restrict__ wqkvT, const h16* __restrict__ wqT,
    const h16* __restrict__ wkT, const h16* __restrict__ wvT,
    const h16* __restrict__ wpT, float* __restrict__ out) {
  extern __shared__ __align__(32) char smem[];
  const int tid  = threadIdx.x;
  const int lane = tid & 31;
  const int w    = tid >> 5;
  const int b    = blockIdx.x;
  const int col  = lane & 15;
  const int g    = (lane >> 4) & 1;

  h16*   sR0  = (h16*)(smem + OFF_R0);
  h16*   sQ   = (h16*)(smem + OFF_Q);
  h16*   sK   = (h16*)(smem + OFF_K);
  h16*   sV   = (h16*)(smem + OFF_V);
  float* scr  = (float*)(smem + OFF_SCR + w * 6144);
  h16*   prob = (h16*)(smem + OFF_SCR + w * 6144 + 4096);

  const float* xg   = x + (size_t)b * (L_TOK * CDIM);
  const float* yg   = y + (size_t)b * (L_TOK * CDIM);
  const float* zg   = z + (size_t)b * (L_TOK * CDIM);
  float*       outg = out + (size_t)b * (L_TOK * CDIM);

  // Prefetch y/z (needed at S4/S5) into L2 while S1-S3 compute.
  for (int i = tid; i < 512; i += 256) {
    __builtin_prefetch(yg + i * 64, 0, 1);
    __builtin_prefetch(zg + i * 64, 0, 1);
  }

  // S0: x -> f16 LDS
  load_conv(xg, sR0, tid);
  __syncthreads();

  // S1: qkv = x @ WqkvT + bqkv  (q pre-scaled; v stored dim-major)
  for (int p = 0; p < 6; ++p) {
    const int ntA = w + 16 * p;
    v8f acc[2][4];
#pragma unroll
    for (int q2 = 0; q2 < 2; ++q2)
#pragma unroll
      for (int i = 0; i < 4; ++i) acc[q2][i] = vzero8();
    for (int kc = 0; kc < 16; ++kc) {
      v16h bfr0 = *(const v16h*)(wqkvT + (ntA * 16 + col) * 512 + kc * 32 + 16 * g);
      v16h bfr1 = *(const v16h*)(wqkvT + ((ntA + 8) * 16 + col) * 512 + kc * 32 + 16 * g);
#pragma unroll
      for (int mt = 0; mt < 4; ++mt) {
        v16h afr   = ldA(sR0, 512, mt * 16 + col, kc * 32, g);
        acc[0][mt] = wmma_(afr, bfr0, acc[0][mt]);
        acc[1][mt] = wmma_(afr, bfr1, acc[1][mt]);
      }
    }
#pragma unroll
    for (int q2 = 0; q2 < 2; ++q2) {
      const int nt  = ntA + 8 * q2;
      float     bb  = bqkv[nt * 16 + col];
      int       s   = nt >> 5;
      int       rem = nt & 31;
      int       hh  = rem >> 1, ph = rem & 1;
      h16*      dst = (h16*)(smem + OFF_Q) + s * 32768;
#pragma unroll
      for (int mt = 0; mt < 4; ++mt) {
#pragma unroll
        for (int r = 0; r < 8; ++r) {
          int   row = mt * 16 + r + 8 * g;
          float v   = acc[q2][mt][r] + bb;
          if (s == 0) v *= QSCALE;
          if (s == 2)  // v: dim-major [head][dim][token]
            dst[(hh * 32 + ph * 16 + col) * 64 + row] = (h16)v;
          else
            dst[(hh * 64 + row) * 32 + ph * 16 + col] = (h16)v;
        }
      }
    }
  }
  __syncthreads();

  // S2: self-attention; xr = x + attn(x) in-place in sR0 (tok-major)
  attn_heads<0>(sQ, sK, sV, sR0, nullptr, scr, prob, w, lane);
  __syncthreads();

  // S3: h = xr @ WpT + bp -> sQ (tok-major f16)
  gemm512<0, 0, false>(sR0, wpT, bp, sQ, nullptr, w, lane);
  __syncthreads();

  // S4: y -> f16 (sK), then k1 = y @ WkT + bk -> head-major into sR0
  load_conv(yg, sK, tid);
  __syncthreads();
  gemm512<0, 1, false>(sK, wkT, bk, sR0, nullptr, w, lane);
  __syncthreads();

  // S5: z -> f16 (sK), then v1 = z @ WvT + bv -> dim-major into sV
  load_conv(zg, sK, tid);
  __syncthreads();
  gemm512<0, 3, false>(sK, wvT, bv, sV, nullptr, w, lane);
  __syncthreads();

  // S6: q1 = (h @ WqT + bq) * SCALE -> head-major into sK
  gemm512<0, 1, true>(sQ, wqT, bq, sK, nullptr, w, lane);
  __syncthreads();

  // S7: cross-attention; x1 = h + attn1 -> head-major into sR0 (over k1)
  attn_heads<1>(sK, sR0, sV, sR0, sQ, scr, prob, w, lane);
  __syncthreads();

  // S8: out = x1 @ WpT + bp -> f32 global
  gemm512<1, 2, false>(sR0, wpT, bp, nullptr, outg, w, lane);
}

// ---------------------------------------------------------------------------
extern "C" void kernel_launch(void* const* d_in, const int* in_sizes, int n_in,
                              void* d_out, int out_size, void* d_ws,
                              size_t ws_size, hipStream_t stream) {
  const float* y    = (const float*)d_in[0];
  const float* z    = (const float*)d_in[1];
  const float* x    = (const float*)d_in[2];
  const float* Wqkv = (const float*)d_in[3];
  const float* bqkv = (const float*)d_in[4];
  const float* Wq   = (const float*)d_in[5];
  const float* bq   = (const float*)d_in[6];
  const float* Wk   = (const float*)d_in[7];
  const float* bk   = (const float*)d_in[8];
  const float* Wv   = (const float*)d_in[9];
  const float* bv   = (const float*)d_in[10];
  const float* Wp   = (const float*)d_in[11];
  const float* bp   = (const float*)d_in[12];

  h16* ws    = (h16*)d_ws;
  h16* wqkvT = ws;                 // 1536*512 halves
  h16* wqT   = wqkvT + 786432;     // 512*512
  h16* wkT   = wqT + 262144;
  h16* wvT   = wkT + 262144;
  h16* wpT   = wvT + 262144;

  wprep_kernel<<<3072, 256, 0, stream>>>(Wqkv, wqkvT, 512, 1536);
  wprep_kernel<<<1024, 256, 0, stream>>>(Wq, wqT, 512, 512);
  wprep_kernel<<<1024, 256, 0, stream>>>(Wk, wkT, 512, 512);
  wprep_kernel<<<1024, 256, 0, stream>>>(Wv, wvT, 512, 512);
  wprep_kernel<<<1024, 256, 0, stream>>>(Wp, wpT, 512, 512);

  fused_mwa_kernel<<<1024, 256, 311296, stream>>>(
      y, z, x, bqkv, bq, bk, bv, bp, wqkvT, wqT, wkT, wvT, wpT, (float*)d_out);
}
